// GPT_Model_38603166057045
// MI455X (gfx1250) — compile-verified, matrix operand order checked by reading, not run on hardware
//
#include <hip/hip_runtime.h>
#include <hip/hip_bf16.h>
#include <cstdint>

// ---------------------------------------------------------------------------
// GPT-2 forward pass for MI455X (gfx1250, wave32, WMMA + TDM).
// GEMMs: bf16 A/B staged in LDS, f32 accumulation via v_wmma_f32_16x16x32_bf16.
// Weight (B) panels streamed with tensor_load_to_lds (TDM, double-buffered).
// Attention is flash-style (no TxT scores buffer).
// ---------------------------------------------------------------------------

typedef __attribute__((ext_vector_type(16))) __bf16 v16bf;
typedef __attribute__((ext_vector_type(8)))  float  v8f;
typedef __attribute__((ext_vector_type(4)))  unsigned int v4u;
typedef __attribute__((ext_vector_type(8)))  int    v8i_;
typedef __attribute__((ext_vector_type(4)))  int    v4i_;

#if defined(__AMDGCN__) && __has_builtin(__builtin_amdgcn_tensor_load_to_lds)
#define HAVE_TDM 1
#else
#define HAVE_TDM 0
#endif

union FragU { uint4 u[2]; v16bf v; };

// Model constants
static constexpr int Bc  = 8;
static constexpr int Tc  = 1024;
static constexpr int Cc  = 1024;
static constexpr int Hc  = 16;
static constexpr int DHc = 64;
static constexpr int Lc  = 8;
static constexpr int FFc = 4096;
static constexpr int Vc  = 50257;
static constexpr int Mrows = Bc * Tc;   // 8192

__device__ __forceinline__ v8f zero8() {
  v8f z;
#pragma unroll
  for (int i = 0; i < 8; ++i) z[i] = 0.0f;
  return z;
}

#if HAVE_TDM
// Issue a TDM 2D tile load: global f32 tensor panel -> LDS (row-major tile).
// D# layout per CDNA5 ISA.
//   tile: rows x cols elements (cols contiguous), row stride = row_stride elems
//   tensor_cols/tensor_rows: remaining extent for hardware OOB zero-fill
__device__ __forceinline__ void tdm_load_2d(unsigned lds_off, const void* gaddr,
                                            unsigned rows, unsigned cols,
                                            unsigned tensor_cols,
                                            unsigned tensor_rows,
                                            unsigned long long row_stride)
{
  const unsigned long long ga = (unsigned long long)(uintptr_t)gaddr;
  v4u g0;
  g0[0] = 1u;                                              // count=1 (valid D#)
  g0[1] = lds_off;                                         // lds_addr [63:32]
  g0[2] = (unsigned)(ga & 0xffffffffu);                    // global_addr[31:0]
  g0[3] = (unsigned)((ga >> 32) & 0x1ffffffu) | (2u << 30);// addr[56:32]|type=2
  v8i_ g1;
  g1[0] = (int)(2u << 16);                                 // data_size=4B, mask=0
  g1[1] = (int)((tensor_cols & 0xffffu) << 16);            // dim0[15:0]
  g1[2] = (int)(((tensor_cols >> 16) & 0xffffu) |
                ((tensor_rows & 0xffffu) << 16));          // dim0[31:16]|dim1[15:0]
  g1[3] = (int)(((tensor_rows >> 16) & 0xffffu) |
                ((cols & 0xffffu) << 16));                 // dim1[31:16]|tile_dim0
  g1[4] = (int)(rows & 0xffffu);                           // tile_dim1 (tile_dim2=0)
  g1[5] = (int)(row_stride & 0xffffffffu);                 // dim0_stride[31:0]
  g1[6] = (int)((row_stride >> 32) & 0xffffu);             // dim0_stride[47:32]
  g1[7] = 0;
  v4i_ z = {0, 0, 0, 0};
#if defined(__clang_major__) && (__clang_major__ >= 23)
  v8i_ z8 = {0, 0, 0, 0, 0, 0, 0, 0};
  __builtin_amdgcn_tensor_load_to_lds(g0, g1, z, z, z8, 0);
#else
  __builtin_amdgcn_tensor_load_to_lds(g0, g1, z, z, 0);
#endif
}
#endif // HAVE_TDM

// Load a 16x32 bf16 fragment from LDS laid out row-major [row][k] with row
// stride `ld`. Matches the CDNA5 16-bit A-matrix layout: lanes 0-15 hold
// K=0..7 & 16..23 of row (lane), lanes 16-31 hold K=8..15 & 24..31.
// The B fragment uses the mirrored layout from LDS holding B^T as [n][k].
__device__ __forceinline__ v16bf ld_frag_lds(const __bf16* base, int ld,
                                             int row16, int k0) {
  const int lane = threadIdx.x & 31;
  const int r  = row16 + (lane & 15);
  const int kh = (lane & 16) ? 8 : 0;
  const __bf16* p = base + r * ld + k0 + kh;
  FragU fu;
  fu.u[0] = *reinterpret_cast<const uint4*>(p);
  fu.u[1] = *reinterpret_cast<const uint4*>(p + 16);
  return fu.v;
}

__device__ __forceinline__ v8f wmma_bf16(v16bf a, v16bf b, v8f c) {
  return __builtin_amdgcn_wmma_f32_16x16x32_bf16(
      /*neg_a=*/false, a, /*neg_b=*/false, b,
      /*c_mod=*/(short)0, c, /*reuse_a=*/false, /*reuse_b=*/false);
}

// ---------------------------------------------------------------------------
// Generic tiled WMMA GEMM: C[M,N] = act(A[M,K] @ W[K,N] + bias) (+ residual)
// BM=128, BK=32, BN = 128 or 64. 256 threads = 8 waves.
//  BN=128: waves in 4x2 grid, each 32x64 (2x4 fragments)
//  BN= 64: waves in 8x1 grid, each 16x64 (1x4 fragments)
// B panels are streamed via TDM (double-buffered f32 staging in LDS), then
// transposed+converted to bf16 for the WMMA B-fragment layout.
// ---------------------------------------------------------------------------
template <int BN>
__global__ void __launch_bounds__(256)
wmma_gemm_kernel(const float* __restrict__ A, const float* __restrict__ Bw,
                 const float* __restrict__ bias, const float* __restrict__ resid,
                 float* __restrict__ Cout, int M, int N, int K,
                 int lda, int ldb, int ldc, int ldres, int do_relu)
{
  constexpr int BM = 128;
  constexpr int BK = 32;
  constexpr int WAVES_N = BN / 64;            // 2 or 1
  constexpr int WV_M = (BN == 128) ? 32 : 16; // rows per wave
  constexpr int FM = WV_M / 16;               // 2 or 1
  constexpr int FN = 4;

  __shared__ alignas(16) __bf16 As[BM * BK];
  __shared__ alignas(16) __bf16 Bt[BN * BK];  // stored transposed: [n][k]
#if HAVE_TDM
  __shared__ alignas(16) float Bstage[2][BK * BN];
#endif

  const int t    = threadIdx.x;
  const int wave = t >> 5;
  const int lane = t & 31;
  const int m0 = blockIdx.y * BM;
  const int n0 = blockIdx.x * BN;

  const int wm = wave / WAVES_N;
  const int wn = wave % WAVES_N;
  const int wrow = wm * WV_M;
  const int wcol = wn * 64;

  v8f acc[FM][FN];
#pragma unroll
  for (int fm = 0; fm < FM; ++fm)
#pragma unroll
    for (int fn = 0; fn < FN; ++fn) acc[fm][fn] = zero8();

  const int ksteps = (K + BK - 1) / BK;

#if HAVE_TDM
  // Prologue: kick off the first B panel DMA.
  if (wave == 0) {
    tdm_load_2d((unsigned)(uintptr_t)&Bstage[0][0],
                Bw + n0, BK, BN,
                (unsigned)(N - n0), (unsigned)K,
                (unsigned long long)ldb);
  }
#endif

  for (int ks = 0; ks < ksteps; ++ks) {
    const int k0 = ks * BK;

    // ---- Stage A tile (BM x BK f32 -> bf16), float4 coalesced ----
#pragma unroll
    for (int i = 0; i < (BM * BK) / (256 * 4); ++i) {
      const int idx4 = i * 256 + t;          // over 1024 float4
      const int row  = idx4 >> 3;            // 8 float4 per row of 32
      const int c4   = (idx4 & 7) * 4;
      const int gm = m0 + row, gk = k0 + c4;
      float4 vv = make_float4(0.f, 0.f, 0.f, 0.f);
      if (gm < M && gk + 3 < K)
        vv = *reinterpret_cast<const float4*>(A + (size_t)gm * lda + gk);
      __bf16* d = As + row * BK + c4;
      d[0] = (__bf16)vv.x; d[1] = (__bf16)vv.y;
      d[2] = (__bf16)vv.z; d[3] = (__bf16)vv.w;
    }

    // Prefetch next A tile into cache while WMMAs run.
    if (ks + 1 < ksteps) {
      const int pr = t >> 1;                 // 0..127
      if (m0 + pr < M)
        __builtin_prefetch(A + (size_t)(m0 + pr) * lda + k0 + BK, 0, 1);
    }

#if HAVE_TDM
    // ---- Pipeline the next B panel DMA, wait for the current one ----
    if (wave == 0) {
      if (ks + 1 < ksteps) {
        tdm_load_2d((unsigned)(uintptr_t)&Bstage[(ks + 1) & 1][0],
                    Bw + (size_t)(k0 + BK) * ldb + n0, BK, BN,
                    (unsigned)(N - n0), (unsigned)(K - (k0 + BK)),
                    (unsigned long long)ldb);
        __builtin_amdgcn_s_wait_tensorcnt(1);
      } else {
        __builtin_amdgcn_s_wait_tensorcnt(0);
      }
    }
    __syncthreads();
    // ---- Convert staged f32 panel -> transposed bf16 Bt[n][k] ----
    {
      const float* bs = &Bstage[ks & 1][0];
#pragma unroll
      for (int i = 0; i < (BK * BN) / 256; ++i) {
        const int idx = i * 256 + t;
        const int kk = idx / BN;
        const int nn = idx % BN;
        Bt[nn * BK + kk] = (__bf16)bs[idx];
      }
    }
#else
    __syncthreads();
    // ---- Fallback: stage B tile directly (BK x BN f32 -> Bt[n][k]) ----
#pragma unroll
    for (int i = 0; i < (BK * BN) / 256; ++i) {
      const int idx = i * 256 + t;
      const int kk = idx / BN;
      const int nn = idx % BN;
      const int gk = k0 + kk, gn = n0 + nn;
      float vv = 0.f;
      if (gk < K && gn < N) vv = Bw[(size_t)gk * ldb + gn];
      Bt[nn * BK + kk] = (__bf16)vv;
    }
#endif
    __syncthreads();

    // ---- Compute: FM x FN WMMA per wave ----
    v16bf afrag[FM];
#pragma unroll
    for (int fm = 0; fm < FM; ++fm)
      afrag[fm] = ld_frag_lds(As, BK, wrow + fm * 16, 0);
#pragma unroll
    for (int fn = 0; fn < FN; ++fn) {
      v16bf bfrag = ld_frag_lds(Bt, BK, wcol + fn * 16, 0);
#pragma unroll
      for (int fm = 0; fm < FM; ++fm)
        acc[fm][fn] = wmma_bf16(afrag[fm], bfrag, acc[fm][fn]);
    }
    __syncthreads();
  }

  // ---- Epilogue: bias + relu + residual, f32 store ----
  const int rowoff = (lane & 16) ? 8 : 0;
  const int ncol   = lane & 15;
#pragma unroll
  for (int fm = 0; fm < FM; ++fm)
#pragma unroll
    for (int fn = 0; fn < FN; ++fn)
#pragma unroll
      for (int r = 0; r < 8; ++r) {
        const int m = m0 + wrow + fm * 16 + rowoff + r;
        const int n = n0 + wcol + fn * 16 + ncol;
        if (m < M && n < N) {
          float c = acc[fm][fn][r];
          if (bias)  c += bias[n];
          if (do_relu) c = fmaxf(c, 0.0f);
          if (resid) c += resid[(size_t)m * ldres + n];
          Cout[(size_t)m * ldc + n] = c;
        }
      }
}

// ---------------------------------------------------------------------------
// Flash attention: one wave per (b, h, 16-query tile). q/k/v/o layout
// [B, T, H*DH] (row stride 1024). Scale 1/sqrt(64) folded into Q.
// ---------------------------------------------------------------------------
__global__ void __launch_bounds__(32)
flash_attn_kernel(const float* __restrict__ q, const float* __restrict__ k,
                  const float* __restrict__ v, float* __restrict__ o)
{
  const int lane = threadIdx.x;
  const int qb = blockIdx.x * 16;
  const int h  = blockIdx.y;
  const int b  = blockIdx.z;

  const size_t rowstride = (size_t)Hc * DHc;     // 1024
  const float* qp = q + (size_t)b * Tc * rowstride + h * DHc;
  const float* kp = k + (size_t)b * Tc * rowstride + h * DHc;
  const float* vp = v + (size_t)b * Tc * rowstride + h * DHc;
  float*       op = o + (size_t)b * Tc * rowstride + h * DHc;

  __shared__ alignas(16) __bf16 Qs[16 * 64];
  __shared__ alignas(16) __bf16 Ks[32 * 64];   // [key][dh]  == B^T for QK^T
  __shared__ alignas(16) __bf16 Vts[64 * 32];  // [dh][key]  == B^T for P*V
  __shared__ alignas(16) __bf16 Ps[16 * 32];   // P tile (A layout source)

  // Load Q tile, fold in softmax scale.
#pragma unroll 4
  for (int i = 0; i < 32; ++i) {
    const int idx = i * 32 + lane;
    const int r = idx >> 6, c = idx & 63;
    Qs[idx] = (__bf16)(qp[(size_t)(qb + r) * rowstride + c] * 0.125f);
  }
  __syncthreads();

  const v16bf aq0 = ld_frag_lds(Qs, 64, 0, 0);
  const v16bf aq1 = ld_frag_lds(Qs, 64, 0, 32);

  const int rowoff = (lane & 16) ? 8 : 0;
  const int ncol   = lane & 15;

  float mrow[8], lrow[8];
#pragma unroll
  for (int r = 0; r < 8; ++r) { mrow[r] = -1e30f; lrow[r] = 0.0f; }
  v8f oacc[4];
#pragma unroll
  for (int nf = 0; nf < 4; ++nf) oacc[nf] = zero8();

  for (int k0 = 0; k0 < qb + 16; k0 += 32) {
    // ---- Stage K (row-major) and V (transposed) tiles as bf16 ----
#pragma unroll 4
    for (int i = 0; i < 64; ++i) {
      const int idx = i * 32 + lane;
      const int r = idx >> 6, c = idx & 63;
      const size_t ga = (size_t)(k0 + r) * rowstride + c;
      Ks[idx]          = (__bf16)kp[ga];
      Vts[c * 32 + r]  = (__bf16)vp[ga];
    }
    __syncthreads();

    // ---- Scores S (16 x 32) = Q (16x64) . K^T ----
    v8f sfr[2];
#pragma unroll
    for (int nb = 0; nb < 2; ++nb) {
      v8f s = zero8();
      s = wmma_bf16(aq0, ld_frag_lds(Ks, 64, nb * 16, 0), s);
      s = wmma_bf16(aq1, ld_frag_lds(Ks, 64, nb * 16, 32), s);
      sfr[nb] = s;
    }

    // ---- Causal mask + online softmax (C-fragment layout) ----
#pragma unroll
    for (int r = 0; r < 8; ++r) {
      const int rowg = qb + rowoff + r;
#pragma unroll
      for (int nb = 0; nb < 2; ++nb) {
        const int colg = k0 + nb * 16 + ncol;
        if (colg > rowg) sfr[nb][r] = -1e30f;
      }
      // row max over the 16 lanes of this half-wave
      float rm = fmaxf(sfr[0][r], sfr[1][r]);
#pragma unroll
      for (int off = 8; off >= 1; off >>= 1)
        rm = fmaxf(rm, __shfl_xor(rm, off));
      const float mnew = fmaxf(mrow[r], rm);
      const float fac  = __expf(mrow[r] - mnew);
      mrow[r] = mnew;
      float rs = 0.0f;
#pragma unroll
      for (int nb = 0; nb < 2; ++nb) {
        sfr[nb][r] = __expf(sfr[nb][r] - mnew);
        rs += sfr[nb][r];
      }
#pragma unroll
      for (int off = 8; off >= 1; off >>= 1)
        rs += __shfl_xor(rs, off);
      lrow[r] = lrow[r] * fac + rs;
#pragma unroll
      for (int nf = 0; nf < 4; ++nf) oacc[nf][r] *= fac;
    }

    // ---- Re-layout P into A-fragment via LDS ----
#pragma unroll
    for (int r = 0; r < 8; ++r)
#pragma unroll
      for (int nb = 0; nb < 2; ++nb)
        Ps[(rowoff + r) * 32 + nb * 16 + ncol] = (__bf16)sfr[nb][r];
    __syncthreads();

    const v16bf pa = ld_frag_lds(Ps, 32, 0, 0);
#pragma unroll
    for (int nf = 0; nf < 4; ++nf) {
      v16bf vb = ld_frag_lds(Vts, 32, nf * 16, 0);
      oacc[nf] = wmma_bf16(pa, vb, oacc[nf]);
    }
    __syncthreads();
  }

  // ---- Normalize and store ----
#pragma unroll
  for (int nf = 0; nf < 4; ++nf)
#pragma unroll
    for (int r = 0; r < 8; ++r) {
      const int mloc = rowoff + r;
      const int d = nf * 16 + ncol;
      op[(size_t)(qb + mloc) * rowstride + d] = oacc[nf][r] / lrow[r];
    }
}

// ---------------------------------------------------------------------------
// Embedding: x[b,t,:] = wte[tok] + wpe[t]
// ---------------------------------------------------------------------------
__global__ void __launch_bounds__(256)
embed_kernel(const int* __restrict__ tokens, const float* __restrict__ wte,
             const float* __restrict__ wpe, float* __restrict__ x)
{
  const int row = blockIdx.x;            // b*T + t
  const int tt  = row % Tc;
  const int tok = tokens[row];
  const float* we = wte + (size_t)tok * Cc;
  const float* wp = wpe + (size_t)tt * Cc;
  float* xr = x + (size_t)row * Cc;
  for (int c = threadIdx.x; c < Cc; c += 256) xr[c] = we[c] + wp[c];
}

// ---------------------------------------------------------------------------
// LayerNorm: one block per row.
// ---------------------------------------------------------------------------
__global__ void __launch_bounds__(256)
layernorm_kernel(const float* __restrict__ x, const float* __restrict__ g,
                 const float* __restrict__ bta, float* __restrict__ out)
{
  const int row = blockIdx.x;
  const float* xr = x + (size_t)row * Cc;
  float s = 0.f, ss = 0.f;
  for (int c = threadIdx.x; c < Cc; c += 256) {
    const float v = xr[c];
    s += v; ss += v * v;
  }
  __shared__ float redA[8], redB[8];
  const int wave = threadIdx.x >> 5, lane = threadIdx.x & 31;
#pragma unroll
  for (int off = 16; off >= 1; off >>= 1) {
    s  += __shfl_down(s, off);
    ss += __shfl_down(ss, off);
  }
  if (lane == 0) { redA[wave] = s; redB[wave] = ss; }
  __syncthreads();
  __shared__ float mu_s, rs_s;
  if (threadIdx.x == 0) {
    float S = 0.f, SS = 0.f;
#pragma unroll
    for (int i = 0; i < 8; ++i) { S += redA[i]; SS += redB[i]; }
    const float mu = S / Cc;
    const float var = SS / Cc - mu * mu;
    mu_s = mu; rs_s = rsqrtf(var + 1e-5f);
  }
  __syncthreads();
  const float mu = mu_s, rs = rs_s;
  float* orow = out + (size_t)row * Cc;
  for (int c = threadIdx.x; c < Cc; c += 256)
    orow[c] = (xr[c] - mu) * rs * g[c] + bta[c];
}

// ---------------------------------------------------------------------------
// Loss: mean over rows of -log_softmax(logits)[target]
// ---------------------------------------------------------------------------
__global__ void loss_init_kernel(float* loss) {
  if (threadIdx.x == 0 && blockIdx.x == 0) *loss = 0.0f;
}

__global__ void __launch_bounds__(256)
loss_kernel(const float* __restrict__ logits, const int* __restrict__ targets,
            float* __restrict__ loss, int Vn, float invN)
{
  const int row = blockIdx.x;
  const float* lr = logits + (size_t)row * Vn;
  const int wave = threadIdx.x >> 5, lane = threadIdx.x & 31;
  __shared__ float red[8];
  __shared__ float bcast;

  float mx = -3e38f;
  for (int i = threadIdx.x; i < Vn; i += 256) mx = fmaxf(mx, lr[i]);
#pragma unroll
  for (int off = 16; off >= 1; off >>= 1) mx = fmaxf(mx, __shfl_xor(mx, off));
  if (lane == 0) red[wave] = mx;
  __syncthreads();
  if (threadIdx.x == 0) {
    float m = red[0];
#pragma unroll
    for (int i = 1; i < 8; ++i) m = fmaxf(m, red[i]);
    bcast = m;
  }
  __syncthreads();
  const float mall = bcast;
  __syncthreads();

  float se = 0.f;
  for (int i = threadIdx.x; i < Vn; i += 256) se += __expf(lr[i] - mall);
#pragma unroll
  for (int off = 16; off >= 1; off >>= 1) se += __shfl_xor(se, off);
  if (lane == 0) red[wave] = se;
  __syncthreads();
  if (threadIdx.x == 0) {
    float S = 0.f;
#pragma unroll
    for (int i = 0; i < 8; ++i) S += red[i];
    const int tgt = targets[row];
    const float lp = lr[tgt] - mall - __logf(S);
    atomicAdd(loss, -lp * invN);
  }
}

// ---------------------------------------------------------------------------
// Host-side launch helpers
// ---------------------------------------------------------------------------
static void gemm128(const float* A, const float* Bw, const float* bias,
                    const float* resid, float* C, int M, int N, int K,
                    int lda, int ldb, int ldc, int ldres, int relu,
                    hipStream_t s)
{
  dim3 g((N + 127) / 128, (M + 127) / 128);
  wmma_gemm_kernel<128><<<g, dim3(256), 0, s>>>(A, Bw, bias, resid, C, M, N, K,
                                                lda, ldb, ldc, ldres, relu);
}

static void gemm64(const float* A, const float* Bw, const float* bias,
                   float* C, int M, int N, int K,
                   int lda, int ldb, int ldc, hipStream_t s)
{
  dim3 g((N + 63) / 64, (M + 127) / 128);
  wmma_gemm_kernel<64><<<g, dim3(256), 0, s>>>(A, Bw, bias, nullptr, C, M, N, K,
                                               lda, ldb, ldc, 0, 0);
}

extern "C" void kernel_launch(void* const* d_in, const int* in_sizes, int n_in,
                              void* d_out, int out_size, void* d_ws, size_t ws_size,
                              hipStream_t stream)
{
  (void)in_sizes; (void)n_in; (void)out_size; (void)ws_size;
  const int*   tokens  = (const int*)d_in[0];
  const int*   targets = (const int*)d_in[1];
  const float* wte   = (const float*)d_in[2];
  const float* wpe   = (const float*)d_in[3];
  const float* ln1_g = (const float*)d_in[4];
  const float* ln1_b = (const float*)d_in[5];
  const float* wq    = (const float*)d_in[6];
  const float* bq    = (const float*)d_in[7];
  const float* wk    = (const float*)d_in[8];
  const float* bk    = (const float*)d_in[9];
  const float* wv    = (const float*)d_in[10];
  const float* bv    = (const float*)d_in[11];
  const float* wo    = (const float*)d_in[12];
  const float* bo    = (const float*)d_in[13];
  const float* ln2_g = (const float*)d_in[14];
  const float* ln2_b = (const float*)d_in[15];
  const float* w1    = (const float*)d_in[16];
  const float* b1    = (const float*)d_in[17];
  const float* w2    = (const float*)d_in[18];
  const float* b2    = (const float*)d_in[19];
  const float* lnf_g = (const float*)d_in[20];
  const float* lnf_b = (const float*)d_in[21];
  const float* w_out = (const float*)d_in[22];
  const float* b_out = (const float*)d_in[23];

  float* logits = (float*)d_out;
  float* loss   = logits + (size_t)Mrows * Vc;

  // Workspace layout (floats)
  const size_t SZ = (size_t)Mrows * Cc;   // 8M floats
  float* ws = (float*)d_ws;
  float* x    = ws;
  float* hbuf = ws + SZ;
  float* qbuf = ws + 2 * SZ;
  float* kbuf = ws + 3 * SZ;
  float* vbuf = ws + 4 * SZ;
  float* obuf = ws + 5 * SZ;
  float* fbuf = ws + 6 * SZ;              // 8192 x 4096

  embed_kernel<<<Mrows, 256, 0, stream>>>(tokens, wte, wpe, x);

  for (int l = 0; l < Lc; ++l) {
    // ln1
    layernorm_kernel<<<Mrows, 256, 0, stream>>>(x, ln1_g + (size_t)l * Cc,
                                                ln1_b + (size_t)l * Cc, hbuf);
    // QKV projections, per head (N = 64)
    for (int hh = 0; hh < Hc; ++hh) {
      const size_t wofs = ((size_t)l * Hc + hh) * Cc * DHc;
      const size_t bofs = ((size_t)l * Hc + hh) * DHc;
      gemm64(hbuf, wq + wofs, bq + bofs, qbuf + hh * DHc,
             Mrows, DHc, Cc, Cc, DHc, Cc, stream);
      gemm64(hbuf, wk + wofs, bk + bofs, kbuf + hh * DHc,
             Mrows, DHc, Cc, Cc, DHc, Cc, stream);
      gemm64(hbuf, wv + wofs, bv + bofs, vbuf + hh * DHc,
             Mrows, DHc, Cc, Cc, DHc, Cc, stream);
    }
    // attention
    flash_attn_kernel<<<dim3(Tc / 16, Hc, Bc), 32, 0, stream>>>(qbuf, kbuf,
                                                                vbuf, obuf);
    // output projection + residual: x = x + o @ wo + bo
    gemm128(obuf, wo + (size_t)l * Cc * Cc, bo + (size_t)l * Cc, x, x,
            Mrows, Cc, Cc, Cc, Cc, Cc, Cc, 0, stream);
    // ln2
    layernorm_kernel<<<Mrows, 256, 0, stream>>>(x, ln2_g + (size_t)l * Cc,
                                                ln2_b + (size_t)l * Cc, hbuf);
    // FFN: f = relu(h @ w1 + b1); x = x + f @ w2 + b2
    gemm128(hbuf, w1 + (size_t)l * Cc * FFc, b1 + (size_t)l * FFc, nullptr,
            fbuf, Mrows, FFc, Cc, Cc, FFc, FFc, 0, 1, stream);
    gemm128(fbuf, w2 + (size_t)l * FFc * Cc, b2 + (size_t)l * Cc, x, x,
            Mrows, Cc, FFc, FFc, Cc, Cc, Cc, 0, stream);
  }

  // final layernorm + logits
  layernorm_kernel<<<Mrows, 256, 0, stream>>>(x, lnf_g, lnf_b, hbuf);
  gemm128(hbuf, w_out, b_out, nullptr, logits,
          Mrows, Vc, Cc, Cc, Vc, Vc, 0, 0, stream);

  // loss
  loss_init_kernel<<<1, 32, 0, stream>>>(loss);
  loss_kernel<<<Mrows, 256, 0, stream>>>(logits, targets, loss, Vc,
                                         1.0f / (float)Mrows);
}
